// constrained_ouroboros_85761906966926
// MI455X (gfx1250) — compile-verified, matrix operand order checked by reading, not run on hardware
//
#include <hip/hip_runtime.h>

// ---- problem constants (from reference setup_inputs) ----
#define B_   8
#define L_   2056
#define LZ   2048       // L - 8
#define DC   256
#define E_   256
#define NS   16         // state dim N
#define KC   4          // conv kernel
#define RR   16         // dt rank
#define NL   2          // layers
#define ROWS (B_*LZ)    // 16384
#define CHUNKS 16
#define CLEN  (LZ/CHUNKS)   // 128

typedef __attribute__((ext_vector_type(16))) __bf16 v16bf;
typedef __attribute__((ext_vector_type(8)))  __bf16 v8bf;
typedef __attribute__((ext_vector_type(8)))  float  v8f;

__device__ __forceinline__ float silu_f(float v) { return v / (1.0f + __expf(-v)); }

// ---------------- generic f32 -> bf16 cast ----------------
__global__ void cast_f32_bf16(const float* __restrict__ src, __bf16* __restrict__ dst, int n)
{
    int i = blockIdx.x * blockDim.x + threadIdx.x;
    if (i < n) dst[i] = (__bf16)src[i];
}

// ---------------- front end: z = [x_c, dxdt] ----------------
__global__ void zder_kernel(const float* __restrict__ x, float* __restrict__ z)
{
    int idx = blockIdx.x * blockDim.x + threadIdx.x;   // over ROWS
    int b = idx >> 11, l = idx & (LZ - 1);
    const float* xb = x + (size_t)b * L_;
    const float c[9] = { 1.f/280.f, -4.f/105.f, 0.2f, -0.8f, 0.f,
                         0.8f, -0.2f, 4.f/105.f, -1.f/280.f };
    float acc = 0.f;
#pragma unroll
    for (int k = 0; k < 9; ++k) acc += c[k] * xb[l + k];
    z[(size_t)idx * 2 + 0] = xb[l + 4];
    z[(size_t)idx * 2 + 1] = acc;
}

// h = [z, flip(z)] @ cp_w.T + cp_b
__global__ void frontproj_kernel(const float* __restrict__ z, const float* __restrict__ w,
                                 const float* __restrict__ bias, float* __restrict__ h)
{
    int idx = blockIdx.x * blockDim.x + threadIdx.x;   // over ROWS*DC
    int d = idx & (DC - 1);
    int row = idx >> 8;
    int b = row >> 11, l = row & (LZ - 1);
    int rf = (b << 11) + (LZ - 1 - l);
    const float* wd = w + d * 4;
    h[idx] = bias[d]
           + z[(size_t)row * 2 + 0] * wd[0] + z[(size_t)row * 2 + 1] * wd[1]
           + z[(size_t)rf  * 2 + 0] * wd[2] + z[(size_t)rf  * 2 + 1] * wd[3];
}

// ---------------- RMSNorm + cast to bf16 (one block per row) ----------------
__global__ void rmsnorm_cast(const float* __restrict__ u, const float* __restrict__ nw,
                             __bf16* __restrict__ out)
{
    int row = blockIdx.x;
    int d = threadIdx.x;
    __shared__ float red[DC];
    float v = u[(size_t)row * DC + d];
    red[d] = v * v;
    __syncthreads();
#pragma unroll
    for (int s = 128; s > 0; s >>= 1) {
        if (d < s) red[d] += red[d + s];
        __syncthreads();
    }
    float scale = rsqrtf(red[0] * (1.0f / DC) + 1e-5f);
    out[(size_t)row * DC + d] = (__bf16)(v * scale * nw[d]);
}

// ---------------- WMMA bf16 GEMM: NT N-tiles per wave, double-buffered K pipe ----
// C[m,n] (+= Cin) = sum_k A[m,k]*W[n,k].  A: MxK (K-major), W: NxK (K-major).
// K is a compile-time constant -> fully unrolled, software-pipelined k-loop so
// next-step fragment loads are in flight while the current NT WMMAs execute.
// Fragment gather per ISA 7.12.2 16-bit layout:
//   lanes 0-15 : elems 0..7 -> K 0..7,  elems 8..15 -> K 16..23
//   lanes 16-31: elems 0..7 -> K 8..15, elems 8..15 -> K 24..31
template<bool ADD_C, int NT, int KDIM>
__global__ void gemm_bf16_wmma(const __bf16* __restrict__ A, const __bf16* __restrict__ W,
                               const float* __restrict__ Cin, float* __restrict__ Cout,
                               int M, int N)
{
    int lane = threadIdx.x & 31;
    int wave = threadIdx.x >> 5;
    int wg   = blockIdx.x * (blockDim.x >> 5) + wave;   // wave-group id
    int groupsN = N / (16 * NT);
    int tm = wg / groupsN;
    int gn = wg - tm * groupsN;
    if (tm * 16 >= M) return;                 // lane-uniform

    int r16 = lane & 15;
    int kh  = (lane >> 4) << 3;               // 0 or 8
    const __bf16* arow = A + (size_t)(tm * 16 + r16) * KDIM + kh;
    int crow0 = tm * 16 + ((lane >> 4) << 3); // C/D: elem r -> row crow0+r

    const __bf16* brow[NT];
    v8f acc[NT];
#pragma unroll
    for (int t = 0; t < NT; ++t) {
        int n0 = (gn * NT + t) * 16;
        brow[t] = W + (size_t)(n0 + r16) * KDIM + kh;
        int ccol = n0 + r16;
#pragma unroll
        for (int r = 0; r < 8; ++r)
            acc[t][r] = ADD_C ? Cin[(size_t)(crow0 + r) * N + ccol] : 0.0f;
    }

    auto frag = [](const __bf16* p) -> v16bf {
        v8bf lo = *(const v8bf*)(p);
        v8bf hi = *(const v8bf*)(p + 16);
        return __builtin_shufflevector(lo, hi, 0,1,2,3,4,5,6,7,8,9,10,11,12,13,14,15);
    };

    // prologue: fragments for k0 = 0
    v16bf av = frag(arow);
    v16bf bv[NT];
#pragma unroll
    for (int t = 0; t < NT; ++t) bv[t] = frag(brow[t]);

#pragma unroll
    for (int k0 = 0; k0 < KDIM; k0 += 32) {
        v16bf av_n;
        v16bf bv_n[NT];
        if (k0 + 32 < KDIM) {                 // compile-time per unrolled iter
            av_n = frag(arow + k0 + 32);
#pragma unroll
            for (int t = 0; t < NT; ++t) bv_n[t] = frag(brow[t] + k0 + 32);
        }
#pragma unroll
        for (int t = 0; t < NT; ++t)
            acc[t] = __builtin_amdgcn_wmma_f32_16x16x32_bf16(false, av, false, bv[t],
                                                             (short)0, acc[t], false, false);
        if (k0 + 32 < KDIM) {
            av = av_n;
#pragma unroll
            for (int t = 0; t < NT; ++t) bv[t] = bv_n[t];
        }
    }
#pragma unroll
    for (int t = 0; t < NT; ++t) {
        int ccol = (gn * NT + t) * 16 + r16;
#pragma unroll
        for (int r = 0; r < 8; ++r)
            Cout[(size_t)(crow0 + r) * N + ccol] = acc[t][r];
    }
}

// ---------------- depthwise causal conv (K=4) + SiLU ----------------
__global__ void conv_silu(const float* __restrict__ xz, const float* __restrict__ cw,
                          const float* __restrict__ cb, float* __restrict__ xc,
                          __bf16* __restrict__ xcb)
{
    int idx = blockIdx.x * blockDim.x + threadIdx.x;   // over ROWS*E
    int e = idx & (E_ - 1);
    int row = idx >> 8;
    int l = row & (LZ - 1);
    int b = row >> 11;
    float a = cb[e];
#pragma unroll
    for (int k = 0; k < KC; ++k) {
        int ls = l - (KC - 1) + k;
        if (ls >= 0)
            a += xz[((size_t)(b * LZ + ls)) * (2 * E_) + e] * cw[e * KC + k];
    }
    a = silu_f(a);
    xc[idx] = a;
    xcb[idx] = (__bf16)a;
}

// ---------------- delta = softplus(dbc[:, :R] @ dw.T + db) ----------------
__global__ void delta_kernel(const float* __restrict__ dbc, const float* __restrict__ dw,
                             const float* __restrict__ db, float* __restrict__ delta)
{
    int idx = blockIdx.x * blockDim.x + threadIdx.x;   // over ROWS*E
    int e = idx & (E_ - 1);
    int row = idx >> 8;
    const float* dr = dbc + (size_t)row * 48;
    const float* wr = dw + e * RR;
    float v = db[e];
#pragma unroll
    for (int r = 0; r < RR; ++r) v += dr[r] * wr[r];
    delta[idx] = (v > 20.f) ? v : log1pf(__expf(v));
}

// ======== chunked selective scan (exact: diagonal linear recurrence) ========
// phase1: per chunk, compute affine map (P = prod dA, S = local scan end state)
__global__ void scan_phase1(const float* __restrict__ delta, const float* __restrict__ xc,
                            const float* __restrict__ dbc, const float* __restrict__ A_log,
                            float* __restrict__ chunkP, float* __restrict__ chunkS)
{
    int bc = blockIdx.x;                 // b*CHUNKS + c
    int b = bc >> 4, c = bc & (CHUNKS - 1);
    int e = threadIdx.x;
    float A[NS], P[NS], S[NS];
#pragma unroll
    for (int n = 0; n < NS; ++n) { A[n] = -__expf(A_log[e * NS + n]); P[n] = 1.f; S[n] = 0.f; }
    size_t base = (size_t)b * LZ + (size_t)c * CLEN;
    for (int l = 0; l < CLEN; ++l) {
        size_t row = base + l;
        if (l + 8 < CLEN) {
            __builtin_prefetch(&delta[(row + 8) * E_ + e], 0, 1);
            __builtin_prefetch(&xc[(row + 8) * E_ + e], 0, 1);
        }
        float dlt = delta[row * E_ + e];
        float dx  = dlt * xc[row * E_ + e];
        const float* bcp = dbc + row * 48 + RR;
#pragma unroll
        for (int n = 0; n < NS; ++n) {
            float dA = __expf(dlt * A[n]);
            S[n] = fmaf(dA, S[n], dx * bcp[n]);
            P[n] *= dA;
        }
    }
    size_t o = ((size_t)bc * E_ + e) * NS;
#pragma unroll
    for (int n = 0; n < NS; ++n) { chunkP[o + n] = P[n]; chunkS[o + n] = S[n]; }
}

// middle: sequentially compose the 16 chunk maps per (b,e,n) -> incoming states
__global__ void scan_mid(const float* __restrict__ chunkP, const float* __restrict__ chunkS,
                         float* __restrict__ hin)
{
    int idx = blockIdx.x * blockDim.x + threadIdx.x;   // over B_*E_*NS = 32768
    int b  = idx / (E_ * NS);
    int en = idx - b * (E_ * NS);
    float h = 0.f;
    for (int c = 0; c < CHUNKS; ++c) {
        size_t o = ((size_t)(b * CHUNKS + c) * E_) * NS + en;
        hin[o] = h;
        h = fmaf(chunkP[o], h, chunkS[o]);
    }
}

// phase2: replay each chunk with the correct incoming state, emit y
__global__ void scan_phase2(const float* __restrict__ delta, const float* __restrict__ xc,
                            const float* __restrict__ dbc, const float* __restrict__ A_log,
                            const float* __restrict__ hin, float* __restrict__ ys)
{
    int bc = blockIdx.x;
    int b = bc >> 4, c = bc & (CHUNKS - 1);
    int e = threadIdx.x;
    float A[NS], h[NS];
    size_t o = ((size_t)bc * E_ + e) * NS;
#pragma unroll
    for (int n = 0; n < NS; ++n) { A[n] = -__expf(A_log[e * NS + n]); h[n] = hin[o + n]; }
    size_t base = (size_t)b * LZ + (size_t)c * CLEN;
    for (int l = 0; l < CLEN; ++l) {
        size_t row = base + l;
        if (l + 8 < CLEN) {
            __builtin_prefetch(&delta[(row + 8) * E_ + e], 0, 1);
            __builtin_prefetch(&xc[(row + 8) * E_ + e], 0, 1);
        }
        float dlt = delta[row * E_ + e];
        float x   = xc[row * E_ + e];
        const float* bcp = dbc + row * 48 + RR;
        float dx = dlt * x;
        float y = 0.f;
#pragma unroll
        for (int n = 0; n < NS; ++n) {
            float dA = __expf(dlt * A[n]);
            h[n] = fmaf(dA, h[n], dx * bcp[n]);
            y = fmaf(h[n], bcp[NS + n], y);
        }
        ys[row * E_ + e] = y;
    }
}

// ---------------- y = (ys + Dp*xc) * silu(zm), cast to bf16 ----------------
__global__ void ycombine(const float* __restrict__ ys, const float* __restrict__ xc,
                         const float* __restrict__ xz, const float* __restrict__ Dp,
                         __bf16* __restrict__ ybf)
{
    int idx = blockIdx.x * blockDim.x + threadIdx.x;   // over ROWS*E
    int e = idx & (E_ - 1);
    int row = idx >> 8;
    float zm = xz[(size_t)row * (2 * E_) + E_ + e];
    float y = (ys[idx] + Dp[e] * xc[idx]) * silu_f(zm);
    ybf[idx] = (__bf16)y;
}

// ---------------- heads: states=flip(h), 4 dot-products, yhat ----------------
__global__ void heads_kernel(const float* __restrict__ h, const float* __restrict__ z,
                             const float* bw, const float* bb,
                             const float* dwv, const float* dbv,
                             const float* omw, const float* omb,
                             const float* gw, const float* gb,
                             const float* dtp,
                             float* __restrict__ yhat, float* __restrict__ states,
                             float* om_s, float* ga_s, float* bb_s, float* dd_s)
{
    int row = blockIdx.x;                     // output row (b,l)
    int b = row >> 11, l = row & (LZ - 1);
    int src = (b << 11) + (LZ - 1 - l);
    int d = threadIdx.x;
    float s = h[(size_t)src * DC + d];
    states[(size_t)row * DC + d] = s;
    __shared__ float r0[DC], r1[DC], r2[DC], r3[DC];
    r0[d] = s * bw[d]; r1[d] = s * dwv[d]; r2[d] = s * omw[d]; r3[d] = s * gw[d];
    __syncthreads();
#pragma unroll
    for (int st = 128; st > 0; st >>= 1) {
        if (d < st) { r0[d] += r0[d+st]; r1[d] += r1[d+st]; r2[d] += r2[d+st]; r3[d] += r3[d+st]; }
        __syncthreads();
    }
    if (d == 0) {
        float bv = fmaxf(r0[0] + bb[0], 0.f) * (1.0f / 1000.0f);
        float dv = fmaxf(r1[0] + dbv[0], 0.f);
        float om = r2[0] + omb[0];
        float ga = (r3[0] + gb[0]) * (1.0f / 1000.0f);
        float z1 = z[(size_t)row * 2 + 0];
        float z2 = z[(size_t)row * 2 + 1] / dtp[0];
        yhat[row] = -om * om * z1 + ga * z2 - bv * z1 * z1 * z2 - dv;
        om_s[row] = om; ga_s[row] = ga; bb_s[row] = bv; dd_s[row] = dv;
    }
}

// ---------------- penalty = mean|d| + (sp(om)+sp(ga)+sp(b))/3 ----------------
__global__ void penalty_kernel(const float* om_s, const float* ga_s,
                               const float* bb_s, const float* dd_s, float* pout)
{
    __shared__ float red[256];
    int t = threadIdx.x;
    float part = 0.f;
    for (int i = t; i < ROWS; i += 256) part += fabsf(dd_s[i]);
    red[t] = part; __syncthreads();
#pragma unroll
    for (int s = 128; s > 0; s >>= 1) { if (t < s) red[t] += red[t + s]; __syncthreads(); }
    float total = red[0] * (1.0f / ROWS);
    __syncthreads();

    const float* arrs[3] = { om_s, ga_s, bb_s };
    float spsum = 0.f;
    for (int q = 0; q < 3; ++q) {
        float spb = 0.f;
        for (int b = 0; b < B_; ++b) {
            const float* a = arrs[q] + (size_t)b * LZ;
            float s1 = 0.f, s2 = 0.f;
            for (int i = 1 + t; i < LZ - 1; i += 256) {  // diffs i=1..2046
                float da = a[i + 1] - a[i];
                s1 += da; s2 += da * da;
            }
            red[t] = s1; __syncthreads();
#pragma unroll
            for (int s = 128; s > 0; s >>= 1) { if (t < s) red[t] += red[t + s]; __syncthreads(); }
            float S1 = red[0]; __syncthreads();
            red[t] = s2; __syncthreads();
#pragma unroll
            for (int s = 128; s > 0; s >>= 1) { if (t < s) red[t] += red[t + s]; __syncthreads(); }
            float S2 = red[0]; __syncthreads();
            float n = (float)(LZ - 2);                   // 2046 samples, ddof=1
            spb += (S2 - S1 * S1 / n) / (n - 1.f);
        }
        spsum += spb * (1.0f / B_);
    }
    if (t == 0) pout[0] = total + spsum * (1.0f / 3.0f);
}

// =====================================================================
extern "C" void kernel_launch(void* const* d_in, const int* in_sizes, int n_in,
                              void* d_out, int out_size, void* d_ws, size_t ws_size,
                              hipStream_t stream)
{
    const float* x        = (const float*)d_in[0];
    const float* dt       = (const float*)d_in[1];
    const float* cp_w     = (const float*)d_in[3];
    const float* cp_b     = (const float*)d_in[4];
    const float* norm_w   = (const float*)d_in[5];
    const float* in_w     = (const float*)d_in[6];
    const float* conv_w   = (const float*)d_in[7];
    const float* conv_b   = (const float*)d_in[8];
    const float* xproj_w  = (const float*)d_in[9];
    const float* dtproj_w = (const float*)d_in[10];
    const float* dtproj_b = (const float*)d_in[11];
    const float* A_log    = (const float*)d_in[12];
    const float* Dp       = (const float*)d_in[13];
    const float* out_w    = (const float*)d_in[14];
    const float* b_w      = (const float*)d_in[15];
    const float* b_b      = (const float*)d_in[16];
    const float* d_w      = (const float*)d_in[17];
    const float* d_b      = (const float*)d_in[18];
    const float* om_w     = (const float*)d_in[19];
    const float* om_b     = (const float*)d_in[20];
    const float* g_w      = (const float*)d_in[21];
    const float* g_b      = (const float*)d_in[22];

    float* out    = (float*)d_out;
    float* yhat   = out;                              // ROWS
    float* states = out + ROWS;                       // ROWS*DC
    float* pen    = out + ROWS + (size_t)ROWS * DC;   // 1

    char* wp = (char*)d_ws;
    auto alloc = [&](size_t bytes) {
        char* p = wp;
        wp += (bytes + 255) & ~(size_t)255;
        return p;
    };
    float*  z     = (float*) alloc((size_t)ROWS * 2 * 4);
    float*  h     = (float*) alloc((size_t)ROWS * DC * 4);
    float*  xz    = (float*) alloc((size_t)ROWS * 2 * E_ * 4);
    float*  xc    = (float*) alloc((size_t)ROWS * E_ * 4);
    float*  dbc   = (float*) alloc((size_t)ROWS * 48 * 4);
    float*  delta = (float*) alloc((size_t)ROWS * E_ * 4);
    float*  ys    = (float*) alloc((size_t)ROWS * E_ * 4);
    __bf16* xnb   = (__bf16*)alloc((size_t)ROWS * DC * 2);
    __bf16* xcb   = (__bf16*)alloc((size_t)ROWS * E_ * 2);
    __bf16* ybf   = (__bf16*)alloc((size_t)ROWS * E_ * 2);
    __bf16* iwb   = (__bf16*)alloc((size_t)NL * 512 * 256 * 2);
    __bf16* xwb   = (__bf16*)alloc((size_t)NL * 48 * 256 * 2);
    __bf16* owb   = (__bf16*)alloc((size_t)NL * 256 * 256 * 2);
    float*  om_s  = (float*) alloc((size_t)ROWS * 4);
    float*  ga_s  = (float*) alloc((size_t)ROWS * 4);
    float*  bb_s  = (float*) alloc((size_t)ROWS * 4);
    float*  dd_s  = (float*) alloc((size_t)ROWS * 4);
    float*  chP   = (float*) alloc((size_t)B_ * CHUNKS * E_ * NS * 4);
    float*  chS   = (float*) alloc((size_t)B_ * CHUNKS * E_ * NS * 4);
    float*  hin   = (float*) alloc((size_t)B_ * CHUNKS * E_ * NS * 4);
    (void)ws_size; (void)in_sizes; (void)n_in; (void)out_size;

    // weight casts (deterministic, every call)
    cast_f32_bf16<<<(NL*512*256 + 255)/256, 256, 0, stream>>>(in_w,    iwb, NL*512*256);
    cast_f32_bf16<<<(NL*48*256  + 255)/256, 256, 0, stream>>>(xproj_w, xwb, NL*48*256);
    cast_f32_bf16<<<(NL*256*256 + 255)/256, 256, 0, stream>>>(out_w,   owb, NL*256*256);

    // front end
    zder_kernel<<<ROWS/256, 256, 0, stream>>>(x, z);
    frontproj_kernel<<<(ROWS*DC)/256, 256, 0, stream>>>(z, cp_w, cp_b, h);

    for (int l = 0; l < NL; ++l) {
        rmsnorm_cast<<<ROWS, 256, 0, stream>>>(h, norm_w + (size_t)l*DC, xnb);

        // in_proj: 16384 x 512 x 256, NT=4 -> 8192 waves
        int waves1 = (ROWS/16) * (512/(16*4));
        gemm_bf16_wmma<false, 4, 256><<<waves1/8, 256, 0, stream>>>(
            xnb, iwb + (size_t)l*512*256, nullptr, xz, ROWS, 512);

        conv_silu<<<(ROWS*E_)/256, 256, 0, stream>>>(
            xz, conv_w + (size_t)l*E_*KC, conv_b + (size_t)l*E_, xc, xcb);

        // x_proj: 16384 x 48 x 256, NT=3 -> 1024 waves
        int waves2 = (ROWS/16) * (48/(16*3));
        gemm_bf16_wmma<false, 3, 256><<<waves2/8, 256, 0, stream>>>(
            xcb, xwb + (size_t)l*48*256, nullptr, dbc, ROWS, 48);

        delta_kernel<<<(ROWS*E_)/256, 256, 0, stream>>>(
            dbc, dtproj_w + (size_t)l*E_*RR, dtproj_b + (size_t)l*E_, delta);

        // chunked scan: 128 parallel chunk blocks instead of 8 serial blocks
        scan_phase1<<<B_*CHUNKS, 256, 0, stream>>>(
            delta, xc, dbc, A_log + (size_t)l*E_*NS, chP, chS);
        scan_mid<<<(B_*E_*NS)/256, 256, 0, stream>>>(chP, chS, hin);
        scan_phase2<<<B_*CHUNKS, 256, 0, stream>>>(
            delta, xc, dbc, A_log + (size_t)l*E_*NS, hin, ys);

        ycombine<<<(ROWS*E_)/256, 256, 0, stream>>>(ys, xc, xz, Dp + (size_t)l*E_, ybf);

        // out_proj + residual: 16384 x 256 x 256, NT=4 -> 4096 waves
        int waves3 = (ROWS/16) * (256/(16*4));
        gemm_bf16_wmma<true, 4, 256><<<waves3/8, 256, 0, stream>>>(
            ybf, owb + (size_t)l*256*256, h, h, ROWS, 256);
    }

    heads_kernel<<<ROWS, 256, 0, stream>>>(h, z, b_w, b_b, d_w, d_b, om_w, om_b,
                                           g_w, g_b, dt, yhat, states,
                                           om_s, ga_s, bb_s, dd_s);
    penalty_kernel<<<1, 256, 0, stream>>>(om_s, ga_s, bb_s, dd_s, pen);
}
